// GANOKernel_14010183319774
// MI455X (gfx1250) — matrix operand order repaired and checked by reading, not compiled, not run on hardware
//
#include <hip/hip_runtime.h>

typedef __attribute__((ext_vector_type(16))) _Float16 v16h;
typedef __attribute__((ext_vector_type(2)))  _Float16 h2;
typedef __attribute__((ext_vector_type(8)))  float    v8f;
typedef __attribute__((ext_vector_type(4)))  float    f4;

#define N_Q    1024
#define N_O    2048
#define LATENT 128
#define HEADS  4
#define QB     4              // queries per block
#define SPLITS 4              // obs-splits per query
#define OTILES (N_O / 16)     // 128 obs tiles of 16
#define TPS    (OTILES / SPLITS)

// ---------------------------------------------------------------------------
// Kernel 1: v = LayerNorm(h_obs) @ Wv + bv   -> ws (f32 [N_O][128])
// one wave per obs row, 8 waves per block
// ---------------------------------------------------------------------------
__global__ __launch_bounds__(256) void value_kernel(
    const float* __restrict__ h_obs, const float* __restrict__ ln_g,
    const float* __restrict__ ln_b,  const float* __restrict__ Wv,
    const float* __restrict__ bv,    float* __restrict__ v)
{
    const int lane = threadIdx.x & 31;
    const int wave = threadIdx.x >> 5;
    const int row  = blockIdx.x * 8 + wave;

    const float* hr = h_obs + row * LATENT;
    f4 x = *(const f4*)(hr + 4 * lane);

    float s1 = x[0] + x[1] + x[2] + x[3];
    float s2 = x[0]*x[0] + x[1]*x[1] + x[2]*x[2] + x[3]*x[3];
    #pragma unroll
    for (int m = 1; m < 32; m <<= 1) {
        s1 += __shfl_xor(s1, m);
        s2 += __shfl_xor(s2, m);
    }
    const float mu   = s1 * (1.0f / LATENT);
    const float var  = s2 * (1.0f / LATENT) - mu * mu;
    const float rstd = rsqrtf(var + 1e-5f);

    f4 g = *(const f4*)(ln_g + 4 * lane);
    f4 b = *(const f4*)(ln_b + 4 * lane);
    f4 hn;
    #pragma unroll
    for (int k = 0; k < 4; ++k) hn[k] = (x[k] - mu) * rstd * g[k] + b[k];

    f4 acc = *(const f4*)(bv + 4 * lane);
    #pragma unroll
    for (int src = 0; src < 32; ++src) {
        #pragma unroll
        for (int k = 0; k < 4; ++k) {
            float hv = __shfl(hn[k], src);           // h_norm[row][4*src+k]
            const int l = src * 4 + k;
            f4 wrow = *(const f4*)(Wv + l * LATENT + 4 * lane);
            acc += hv * wrow;
        }
    }
    *(f4*)(v + row * LATENT + 4 * lane) = acc;
}

// ---------------------------------------------------------------------------
// Kernel 2: fused pair-MLP (WMMA f16) + online softmax + aggregation
// block = 512 threads = 16 waves = QB queries x SPLITS obs-splits
// ---------------------------------------------------------------------------
__global__ __launch_bounds__(512, 1) void gano_attn(
    const float* __restrict__ pos_obs,   const float* __restrict__ pos_query,
    const int*   __restrict__ obs_batch, const int*   __restrict__ query_batch,
    const float* __restrict__ W1, const float* __restrict__ b1,
    const float* __restrict__ W2, const float* __restrict__ b2,
    const float* __restrict__ v, float* __restrict__ out)
{
    // W1 augmented: rows k=0..9 = W1, k=10 = b1 (kin[10]==1), k>=11 zero.
    __shared__ __align__(16) _Float16 sW1[LATENT][16];         // [n][k]   4 KB
    __shared__ __align__(16) _Float16 sW2[16][LATENT];         // [h][n]   4 KB (h>=4 zero)
    __shared__ __align__(16) float    sV[SPLITS][16 * LATENT]; // 32 KB
    __shared__ __align__(16) float    sComb[QB][SPLITS][136];  // 8.5 KB

    const int tid   = threadIdx.x;
    const int lane  = tid & 31;
    const int wave  = tid >> 5;
    const int split = wave & (SPLITS - 1);
    const int ql    = wave >> 2;
    const int q     = blockIdx.x * QB + ql;
    const int half  = lane >> 4;     // WMMA lane half
    const int n16   = lane & 15;
    const int hl    = lane >> 3;     // this lane's head (acc dims 4*lane..+3)
    const float NEG_INF = -__builtin_inff();

    // ---- cooperative weight staging -------------------------------------
    for (int i = tid; i < LATENT * 16; i += 512) {
        int n = i >> 4, k = i & 15;
        float val = (k < 10) ? W1[k * LATENT + n] : ((k == 10) ? b1[n] : 0.0f);
        sW1[n][k] = (_Float16)val;
    }
    for (int i = tid; i < 16 * LATENT; i += 512) {
        int h = i >> 7, n = i & 127;
        float val = (h < HEADS) ? W2[n * HEADS + h] : 0.0f;
        sW2[h][n] = (_Float16)val;
    }
    __syncthreads();

    // ---- per-lane resident fragments ------------------------------------
    // A1[t] = W1aug^T tile t (rows n = 16t..16t+15, K = 11 real of 32)
    v16h A1[8];
    #pragma unroll
    for (int t = 0; t < 8; ++t) {
        int n = t * 16 + n16;
        #pragma unroll
        for (int j = 0; j < 8; ++j) {
            int K = 2 * j;                            // half 0: K = s (<16, rows >=11 zero); half 1 (K >= 16) zero
            h2 p = *(const h2*)&sW1[n][K];
            if (half) { p[0] = (_Float16)0.0f; p[1] = (_Float16)0.0f; }
            A1[t][2 * j] = p[0]; A1[t][2 * j + 1] = p[1];
        }
    }
    // B2[c] = W2 K-chunk c (K = n, N col = head)
    v16h B2[4];
    #pragma unroll
    for (int c = 0; c < 4; ++c)
        #pragma unroll
        for (int j = 0; j < 8; ++j) {
            int K = c * 32 + (j >> 2) * 16 + half * 8 + 2 * (j & 3);
            h2 p = *(const h2*)&sW2[n16][K];
            B2[c][2 * j] = p[0]; B2[c][2 * j + 1] = p[1];
        }

    const float pqx = pos_query[q * 3 + 0];
    const float pqy = pos_query[q * 3 + 1];
    const float pqz = pos_query[q * 3 + 2];
    const int   qb  = query_batch[q];
    const float b2l = b2[hl];

    float m_run = NEG_INF, s_run = 0.0f;
    f4 acc = {0.0f, 0.0f, 0.0f, 0.0f};
    const v8f z8 = {};
    const h2  z2 = {(_Float16)0.0f, (_Float16)0.0f};

    // async-staging addresses for this lane (covers the split's 8 KB v tile)
    const int      gidx    = ql * 32 + lane;                    // 0..127
    const unsigned ldsbase = (unsigned)(uintptr_t)&sV[split][0];

    for (int tt = 0; tt < TPS; ++tt) {
        const int tile = tt * SPLITS + split;
        const int o0   = tile * 16;

        // ---- launch async global->LDS stage of the v tile (ASYNCcnt).
        // Safe: previous end-of-iteration barrier certified all readers done.
        // The wait is deferred until just before aggregation, so the whole
        // WMMA + softmax-prep section hides the staging latency.
        {
            const char* gsrc = (const char*)(v + o0 * LATENT);
            #pragma unroll
            for (int u = 0; u < 4; ++u) {
                unsigned    loff = ldsbase + 16u * (unsigned)(gidx + 128 * u);
                const void* gp   = gsrc + 16 * (gidx + 128 * u);
                asm volatile("global_load_async_to_lds_b128 %0, %1, off"
                             :: "v"(loff), "v"(gp) : "memory");
            }
        }

        // ---- build kin^T B-fragment (pair column = n16) -----------------
        const int   o   = o0 + n16;
        const float pox = pos_obs[o * 3 + 0];
        const float poy = pos_obs[o * 3 + 1];
        const float poz = pos_obs[o * 3 + 2];
        const float rx = pqx - pox, ry = pqy - poy, rz = pqz - poz;
        const float dist = __builtin_amdgcn_sqrtf(rx * rx + ry * ry + rz * rz);
        const int   ob   = obs_batch[o];
        const unsigned mask16 = (unsigned)__ballot(ob == qb) & 0xFFFFu;

        const float zm = (half == 0) ? 1.0f : 0.0f;       // K=16..31 absent
        v16h KB;
        KB[0] = (_Float16)(pqx * zm); KB[1] = (_Float16)(pqy * zm);
        KB[2] = (_Float16)(pqz * zm); KB[3] = (_Float16)(pox * zm);
        KB[4] = (_Float16)(poy * zm); KB[5] = (_Float16)(poz * zm);
        KB[6] = (_Float16)(rx * zm);  KB[7] = (_Float16)(ry * zm);
        KB[8] = (_Float16)(rz * zm);  KB[9] = (_Float16)(dist * zm);
        KB[10] = (_Float16)zm;                             // bias-1 (b1 row)
        #pragma unroll
        for (int e = 11; e < 16; ++e) KB[e] = (_Float16)0.0f;

        // ---- layer 1 (transposed): 8 independent WMMAs back-to-back -----
        v8f C1[8];
        #pragma unroll
        for (int t = 0; t < 8; ++t)
            C1[t] = __builtin_amdgcn_wmma_f32_16x16x32_f16(
                        false, A1[t], false, KB, (short)0, z8, false, false);

        // ---- packed relu -> layer-2 A fragments (lane-local relayout) ---
        v16h A2[4];
        #pragma unroll
        for (int c = 0; c < 4; ++c) {
            #pragma unroll
            for (int i = 0; i < 4; ++i) {
                h2 pa; pa[0] = (_Float16)C1[2*c][2*i];   pa[1] = (_Float16)C1[2*c][2*i+1];
                pa = __builtin_elementwise_max(pa, z2);        // v_pk_max_num_f16
                A2[c][2*i] = pa[0]; A2[c][2*i+1] = pa[1];
                h2 pb; pb[0] = (_Float16)C1[2*c+1][2*i]; pb[1] = (_Float16)C1[2*c+1][2*i+1];
                pb = __builtin_elementwise_max(pb, z2);
                A2[c][8+2*i] = pb[0]; A2[c][8+2*i+1] = pb[1];
            }
        }

        // ---- layer 2: logits[m][h], K = 128 over 4 chained WMMAs --------
        v8f D2 = z8;
        #pragma unroll
        for (int c = 0; c < 4; ++c)
            D2 = __builtin_amdgcn_wmma_f32_16x16x32_f16(
                     false, A2[c], false, B2[c], (short)0, D2, false, false);

        // ---- extract this lane's head column: logit[o][hl] --------------
        float Lo[16];
        #pragma unroll
        for (int r = 0; r < 8; ++r) {
            Lo[r]     = __shfl(D2[r], hl);        // m = r      (lanes 0..15)
            Lo[r + 8] = __shfl(D2[r], hl + 16);   // m = r + 8  (lanes 16..31)
        }
        float tmax = NEG_INF;
        #pragma unroll
        for (int o16 = 0; o16 < 16; ++o16) {
            float lg = ((mask16 >> o16) & 1u) ? (Lo[o16] + b2l) : NEG_INF;
            Lo[o16] = lg;
            tmax = fmaxf(tmax, lg);
        }
        const float newm = fmaxf(m_run, tmax);

        // ---- staged v tile now needed: drain ASYNCcnt, make visible -----
        asm volatile("s_wait_asynccnt 0x0" ::: "memory");
        __syncthreads();

        // ---- online softmax + aggregation -------------------------------
        if (newm > NEG_INF) {
            const float alpha = __expf(m_run - newm);
            s_run *= alpha;
            acc   *= alpha;
            float wsum = 0.0f;
            #pragma unroll
            for (int o16 = 0; o16 < 16; ++o16) {
                float w = __expf(Lo[o16] - newm);      // masked -> exp(-inf)=0
                wsum += w;
                f4 vv = *(const f4*)&sV[split][o16 * LATENT + 4 * lane];
                acc += w * vv;
            }
            s_run += wsum;
            m_run = newm;
        }
        __syncthreads();                 // all readers done -> next stage OK
    }

    // ---- combine the 4 obs-splits of each query -------------------------
    {
        float* cb = &sComb[ql][split][0];
        *(f4*)(cb + 4 * lane) = acc;
        if ((lane & 7) == 0) { cb[128 + hl] = m_run; cb[132 + hl] = s_run; }
    }
    __syncthreads();
    if (split == 0) {
        float M = NEG_INF;
        #pragma unroll
        for (int s = 0; s < SPLITS; ++s) M = fmaxf(M, sComb[ql][s][128 + hl]);
        float T = 0.0f;
        f4 o4 = {0.0f, 0.0f, 0.0f, 0.0f};
        #pragma unroll
        for (int s = 0; s < SPLITS; ++s) {
            float ms = sComb[ql][s][128 + hl];
            float ss = sComb[ql][s][132 + hl];
            float sc = (ms > NEG_INF) ? __expf(ms - M) : 0.0f;
            T += ss * sc;
            f4 a = *(const f4*)&sComb[ql][s][4 * lane];
            o4 += sc * a;
        }
        f4 res = o4 * (1.0f / T);
        *(f4*)&out[q * LATENT + 4 * lane] = res;
    }
}

// ---------------------------------------------------------------------------
extern "C" void kernel_launch(void* const* d_in, const int* in_sizes, int n_in,
                              void* d_out, int out_size, void* d_ws, size_t ws_size,
                              hipStream_t stream) {
    const float* h_obs       = (const float*)d_in[0];
    const float* pos_obs     = (const float*)d_in[1];
    const float* pos_query   = (const float*)d_in[2];
    const int*   obs_batch   = (const int*)  d_in[3];
    const int*   query_batch = (const int*)  d_in[4];
    const float* W1          = (const float*)d_in[5];
    const float* b1          = (const float*)d_in[6];
    const float* W2          = (const float*)d_in[7];
    const float* b2          = (const float*)d_in[8];
    const float* ln_g        = (const float*)d_in[9];
    const float* ln_b        = (const float*)d_in[10];
    const float* Wv          = (const float*)d_in[11];
    const float* bv          = (const float*)d_in[12];
    float* out = (float*)d_out;
    float* v   = (float*)d_ws;                    // f32 [N_O][LATENT] = 1 MB

    value_kernel<<<N_O / 8, 256, 0, stream>>>(h_obs, ln_g, ln_b, Wv, bv, v);
    gano_attn<<<N_Q / QB, 512, 0, stream>>>(pos_obs, pos_query, obs_batch,
                                            query_batch, W1, b1, W2, b2, v, out);
}
